// FlowSeekCorrelationPyramid_40905268527527
// MI455X (gfx1250) — matrix operand order repaired and checked by reading, not compiled, not run on hardware
//
#include <hip/hip_runtime.h>
#include <hip/hip_bf16.h>
#include <math.h>

typedef __attribute__((ext_vector_type(16))) _Float16 v16h;
typedef __attribute__((ext_vector_type(8)))  float    v8f;
typedef __attribute__((ext_vector_type(4)))  int      v4i;

#define NB 4
#define NC 256
#define NH 128
#define NW 192
#define RAD 4

union FragU { v4i i[2]; v16h h; };

// ---------------------------------------------------------------------------
// Pool (s x s mean) + L2-normalize along C, write f16 pixel-major [pix][256].
// One wave (32 lanes) per output pixel; each lane owns 8 contiguous channels
// so the f16 store is a contiguous 16B run per lane (coalesced 512B/wave).
// ---------------------------------------------------------------------------
__global__ __launch_bounds__(256)
void prep_feat(const float* __restrict__ src, _Float16* __restrict__ dst,
               int s, int h, int w, int npix)
{
    int wid  = blockIdx.x * (blockDim.x >> 5) + (threadIdx.x >> 5);
    int lane = threadIdx.x & 31;
    if (wid >= npix) return;

    int x = wid % w;
    int y = (wid / w) % h;
    int b = wid / (w * h);

    float vals[8];
    float ss = 0.f;
    const float inv_cnt = 1.0f / (float)(s * s);

    #pragma unroll
    for (int i = 0; i < 8; ++i) {
        int c = lane * 8 + i;
        const float* p = src + (((size_t)b * NC + c) * NH + (size_t)y * s) * NW + (size_t)x * s;
        float acc = 0.f;
        for (int yy = 0; yy < s; ++yy) {
            const float* pr = p + (size_t)yy * NW;
            for (int xx = 0; xx < s; ++xx) acc += pr[xx];
        }
        vals[i] = acc * inv_cnt;
        ss += vals[i] * vals[i];
    }
    // wave32 reduction of sum of squares
    #pragma unroll
    for (int off = 16; off >= 1; off >>= 1) ss += __shfl_xor(ss, off, 32);

    float inv = 1.0f / fmaxf(sqrtf(ss), 1e-12f);

    _Float16* o = dst + ((size_t)wid) * NC;
    #pragma unroll
    for (int i = 0; i < 8; ++i)
        o[lane * 8 + i] = (_Float16)(vals[i] * inv);
}

// ---------------------------------------------------------------------------
// Pooled depth bonus: dt[b,y,x] = w * exp(-|pool(d1) - pool(d2)|)
// ---------------------------------------------------------------------------
__global__ __launch_bounds__(256)
void prep_depth(const float* __restrict__ d1, const float* __restrict__ d2,
                const float* __restrict__ dwp, float* __restrict__ dt,
                int s, int h, int w, int npix)
{
    int idx = blockIdx.x * blockDim.x + threadIdx.x;
    if (idx >= npix) return;
    int x = idx % w;
    int y = (idx / w) % h;
    int b = idx / (w * h);

    const float inv_cnt = 1.0f / (float)(s * s);
    size_t base = ((size_t)b * NH + (size_t)y * s) * NW + (size_t)x * s;
    float a1 = 0.f, a2 = 0.f;
    for (int yy = 0; yy < s; ++yy) {
        const float* p1 = d1 + base + (size_t)yy * NW;
        const float* p2 = d2 + base + (size_t)yy * NW;
        for (int xx = 0; xx < s; ++xx) { a1 += p1[xx]; a2 += p2[xx]; }
    }
    a1 *= inv_cnt; a2 *= inv_cnt;
    dt[idx] = dwp[0] * expf(-fabsf(a1 - a2));
}

// ---------------------------------------------------------------------------
// WMMA correlation. One wave per (b, y, 16-wide x-tile).
// A: 16 f1 pixels x K=256 (preloaded once, reused for all 18 tile-GEMMs).
// B: 32-wide f2 band at row y-dy, columns x0-8 .. x0+23 (2 N-tiles).
//    Addresses are clamped in-bounds; out-of-image columns are zeroed
//    branch-free with a per-lane AND mask (zero-pad semantics).
// D[m,n] = f1(x0+m) . f2(x0-8+n); valid displacement dx = m - n + 8.
//
// Fragment layouts per CDNA5 ISA 7.12.2 (wave32):
//   A 16x32 f16 : lane<16 -> row=lane,   halves {K0..7, K16..23}
//                 lane>=16 -> row=lane-16, halves {K8..15, K24..31}
//   B 32x16 f16 : lane<16 -> col=lane,   halves {K0..7, K16..23}
//                 lane>=16 -> col=lane-16, halves {K8..15, K24..31}
//   C/D 16x16 f32: lane<16 -> M=vgpr, N=lane ; lane>=16 -> M=8+vgpr, N=lane-16
// ---------------------------------------------------------------------------
__global__ __launch_bounds__(32)
void corr_wmma(const _Float16* __restrict__ f1, const _Float16* __restrict__ f2,
               const float* __restrict__ dt, float* __restrict__ out,
               int h, int w)
{
    const int lane    = threadIdx.x;
    const int tiles_x = w >> 4;
    const int tx = blockIdx.x % tiles_x;
    const int y  = (blockIdx.x / tiles_x) % h;
    const int b  =  blockIdx.x / (tiles_x * h);
    const int x0 = tx << 4;

    // ---- preload A fragments (8 K-steps of 32), 2 x b128 each ----
    const int mrow  = lane & 15;
    const int koffA = (lane >> 4) * 8;
    const _Float16* a_base = f1 + (((size_t)b * h + y) * w + x0 + mrow) * NC + koffA;

    v16h A[8];
    #pragma unroll
    for (int ks = 0; ks < 8; ++ks) {
        FragU u;
        u.i[0] = *(const v4i*)(a_base + ks * 32);        // K +0..7
        u.i[1] = *(const v4i*)(a_base + ks * 32 + 16);   // K +16..23
        A[ks] = u.h;
    }

    const int ncol  = lane & 15;          // column within 16-wide N-tile
    const int koffB = (lane >> 4) * 8;    // K 8-group selected by lane group
    const int mm_hi = (lane >> 4) << 3;   // +8 rows for upper lane half (D layout)

    // ---- depth bonus for this lane's 8 output pixels (reused everywhere) ----
    const float* dt_row = dt + ((size_t)b * h + y) * w;
    float dtv[8];
    #pragma unroll
    for (int r = 0; r < 8; ++r) dtv[r] = dt_row[x0 + r + mm_hi];

    for (int dy = -RAD; dy <= RAD; ++dy) {
        const int ys = y - dy;
        const bool rowok = (ys >= 0) && (ys < h);
        const int ysc = rowok ? ys : 0;

        for (int nt = 0; nt < 2; ++nt) {
            const int gx = x0 - 8 + ncol + nt * 16;      // f2 column for this lane
            const bool ok = rowok && (gx >= 0) && (gx < w);
            const int gxc = (gx < 0) ? 0 : ((gx >= w) ? (w - 1) : gx);
            const int msk = ok ? ~0 : 0;
            const v4i msk4 = { msk, msk, msk, msk };
            const _Float16* b_base = f2 + (((size_t)b * h + ysc) * w + gxc) * NC + koffB;

            // batch-load all 8 B fragments (16 x global_load_b128), mask, then MMA
            v16h Bf[8];
            #pragma unroll
            for (int ks = 0; ks < 8; ++ks) {
                FragU u;
                u.i[0] = *(const v4i*)(b_base + ks * 32) & msk4;      // K +0..7
                u.i[1] = *(const v4i*)(b_base + ks * 32 + 16) & msk4; // K +16..23
                Bf[ks] = u.h;
            }

            v8f acc = {};
            #pragma unroll
            for (int ks = 0; ks < 8; ++ks)
                acc = __builtin_amdgcn_wmma_f32_16x16x32_f16(
                        false, A[ks], false, Bf[ks], (short)0, acc, false, false);

            // ---- scatter diagonal bands to the 81 output channels ----
            const int nn = ncol + nt * 16;               // 0..31 within band
            #pragma unroll
            for (int r = 0; r < 8; ++r) {
                const int mm = r + mm_hi;                // output pixel m in tile
                const int dx = mm - nn + 8;
                if (dx >= -RAD && dx <= RAD) {
                    const int x = x0 + mm;
                    const int disp = (dy + RAD) * (2 * RAD + 1) + (dx + RAD);
                    const size_t oidx = (((size_t)b * 81 + disp) * h + y) * w + x;
                    out[oidx] = acc[r] + dtv[r];
                }
            }
        }
    }
}

extern "C" void kernel_launch(void* const* d_in, const int* in_sizes, int n_in,
                              void* d_out, int out_size, void* d_ws, size_t ws_size,
                              hipStream_t stream)
{
    const float* fmap1  = (const float*)d_in[0];
    const float* fmap2  = (const float*)d_in[1];
    const float* depth1 = (const float*)d_in[2];
    const float* depth2 = (const float*)d_in[3];
    const float* dwp    = (const float*)d_in[4];
    float* out = (float*)d_out;

    // workspace: level-0-sized buffers, reused for every level (stream-ordered)
    const size_t maxFeat = (size_t)NB * NH * NW * NC;     // elements (f16)
    _Float16* f1f = (_Float16*)d_ws;
    _Float16* f2f = f1f + maxFeat;
    float*    dt  = (float*)(f2f + maxFeat);

    size_t out_off = 0;
    for (int lvl = 0; lvl < 4; ++lvl) {
        const int s = 1 << lvl;
        const int h = NH >> lvl;
        const int w = NW >> lvl;
        const int npix = NB * h * w;

        const int fblocks = (npix + 7) / 8;               // 8 waves / block
        prep_feat<<<fblocks, 256, 0, stream>>>(fmap1, f1f, s, h, w, npix);
        prep_feat<<<fblocks, 256, 0, stream>>>(fmap2, f2f, s, h, w, npix);
        prep_depth<<<(npix + 255) / 256, 256, 0, stream>>>(depth1, depth2, dwp, dt,
                                                           s, h, w, npix);

        const int cblocks = NB * h * (w >> 4);            // one wave each
        corr_wmma<<<cblocks, 32, 0, stream>>>(f1f, f2f, dt, out + out_off, h, w);

        out_off += (size_t)NB * 81 * h * w;
    }
}
